// QuantBasicTransformerBlock_8211977470500
// MI455X (gfx1250) — compile-verified, matrix operand order checked by reading, not run on hardware
//
#include <hip/hip_runtime.h>
#include <math.h>

// ---------------------------------------------------------------------------
// CDNA5 (gfx1250) wave32 WMMA transformer block.
// All matmuls via v_wmma_f32_16x16x32_f16 (f16 operands, fp32 accumulate).
// GEMM staging uses CDNA5 async DMA: global_load_async_to_lds_b128 (ASYNCcnt),
// double-buffered in LDS; fragments via ds_load_b128; guard-free (all shapes
// padded to tile multiples).
// Main GEMM: 128x128 block, 8 waves, 32x64 per wave (8 WMMA / k-step).
// P@V GEMM : 128x64 block, K-major B staging (LDS transpose), 32x32 per wave.
// ---------------------------------------------------------------------------

typedef __attribute__((ext_vector_type(16))) _Float16 v16h;
typedef __attribute__((ext_vector_type(8)))  float    v8f;

#define BM 128
#define BN 128
#define BNPV 64
#define BK 32
#define LDS_STRIDE 40   // BK + 8 halfs pad (80B row, keeps 16B alignment)

enum { EPI_F32 = 0, EPI_FQ = 1, EPI_F16 = 2, EPI_BIAS_F16 = 3, EPI_BIAS_RES_F32 = 4 };

// ------------------- CDNA5 async global->LDS DMA helpers -------------------
// GLOBAL_LOAD_ASYNC_TO_LDS_B128: per-lane copy of 16B memory -> LDS, tracked
// by ASYNCcnt (ISA 10.x / 15.18.3). LDS dest passed as 32-bit LDS address.
__device__ __forceinline__ void async_b128(unsigned lds_off, const void* gptr) {
    asm volatile("global_load_async_to_lds_b128 %0, %1, off"
                 :: "v"(lds_off), "v"(gptr)
                 : "memory");
}
__device__ __forceinline__ void wait_async0() {
    asm volatile("s_wait_asynccnt 0x0" ::: "memory");
}
__device__ __forceinline__ unsigned lds_addr(const void* p) {
    return (unsigned)(size_t)p;   // LDS aperture: offset lives in low 32 bits
}

// -------------------------- fragment load helper ---------------------------
// 16-bit A/B operand layout (ISA 7.12.2): lanes 0-15 row M=l16 K chunks
// [0..7],[16..23]; lanes 16-31 row M=l16 K chunks [8..15],[24..31].
__device__ __forceinline__ v16h load_frag(const _Float16* rowbase, int hi) {
    union { v16h v; int4 q[2]; } u;
    u.q[0] = *(const int4*)(rowbase + hi * 8);
    u.q[1] = *(const int4*)(rowbase + hi * 8 + 16);
    return u.v;
}

// ------------------------------- main GEMM ---------------------------------
// C[M,N] = A[M,K] (f16, lda) @ Bt[N][K] (f16, ldb).
// Requires M%128==0, N%128==0, K%32==0.
template <int EPI>
__global__ __launch_bounds__(256) void k_gemm(
    const _Float16* __restrict__ A, int lda,
    const _Float16* __restrict__ B, int ldb,
    float* __restrict__ Cf, _Float16* __restrict__ Ch, int ldc,
    int M, int N, int K,
    const float* __restrict__ bias,
    const float* __restrict__ resid, int ldr,
    const float* __restrict__ s_delta, const float* __restrict__ s_zp,
    float alpha)
{
    __shared__ alignas(16) _Float16 As[2][BM][LDS_STRIDE];
    __shared__ alignas(16) _Float16 Bs[2][BN][LDS_STRIDE];

    const int tid  = threadIdx.x;
    const int lane = tid & 31;
    const int wid  = tid >> 5;           // 8 waves
    const int l16  = lane & 15;
    const int hi   = (lane >> 4) & 1;
    const int wm   = wid & 3;            // wave M tile (4 x 32 rows)
    const int wn   = wid >> 2;           // wave N tile (2 x 64 cols)
    const int bm0  = blockIdx.x * BM;
    const int bn0  = blockIdx.y * BN;

    v8f acc[2][4] = {};

    // A and B: 512 chunks of 8 halfs each (2/thread each).
    const int cr  = tid >> 2;          // 0..63 (rows; +64 for 2nd chunk)
    const int ckc = (tid & 3) * 8;     // k offset
    const _Float16* Arow0 = A + (size_t)(bm0 + cr) * lda + ckc;
    const _Float16* Arow1 = A + (size_t)(bm0 + cr + 64) * lda + ckc;
    const _Float16* Brow0 = B + (size_t)(bn0 + cr) * ldb + ckc;
    const _Float16* Brow1 = B + (size_t)(bn0 + cr + 64) * ldb + ckc;

    unsigned lA0[2], lA1[2], lB0[2], lB1[2];
    #pragma unroll
    for (int s = 0; s < 2; ++s) {
        lA0[s] = lds_addr(&As[s][cr][ckc]);
        lA1[s] = lds_addr(&As[s][cr + 64][ckc]);
        lB0[s] = lds_addr(&Bs[s][cr][ckc]);
        lB1[s] = lds_addr(&Bs[s][cr + 64][ckc]);
    }

    // prologue: DMA tile 0 into buffer 0
    async_b128(lA0[0], Arow0);
    async_b128(lA1[0], Arow1);
    async_b128(lB0[0], Brow0);
    async_b128(lB1[0], Brow1);
    wait_async0();
    __syncthreads();

    int buf = 0;
    for (int k0 = BK; k0 < K; k0 += BK) {
        int nb = buf ^ 1;
        // kick off DMA of next tile into the other buffer (overlaps WMMAs)
        async_b128(lA0[nb], Arow0 + k0);
        async_b128(lA1[nb], Arow1 + k0);
        async_b128(lB0[nb], Brow0 + k0);
        async_b128(lB1[nb], Brow1 + k0);
        // compute current tile: 2x4 fragments -> 8 WMMA
        {
            v16h afr[2], bfr[4];
            #pragma unroll
            for (int i = 0; i < 2; ++i)
                afr[i] = load_frag(&As[buf][wm * 32 + i * 16 + l16][0], hi);
            #pragma unroll
            for (int j = 0; j < 4; ++j)
                bfr[j] = load_frag(&Bs[buf][wn * 64 + j * 16 + l16][0], hi);
            #pragma unroll
            for (int i = 0; i < 2; ++i)
                #pragma unroll
                for (int j = 0; j < 4; ++j)
                    acc[i][j] = __builtin_amdgcn_wmma_f32_16x16x32_f16(
                        false, afr[i], false, bfr[j], (short)0, acc[i][j], false, false);
        }
        wait_async0();
        __syncthreads();
        buf = nb;
    }
    {
        v16h afr[2], bfr[4];
        #pragma unroll
        for (int i = 0; i < 2; ++i)
            afr[i] = load_frag(&As[buf][wm * 32 + i * 16 + l16][0], hi);
        #pragma unroll
        for (int j = 0; j < 4; ++j)
            bfr[j] = load_frag(&Bs[buf][wn * 64 + j * 16 + l16][0], hi);
        #pragma unroll
        for (int i = 0; i < 2; ++i)
            #pragma unroll
            for (int j = 0; j < 4; ++j)
                acc[i][j] = __builtin_amdgcn_wmma_f32_16x16x32_f16(
                    false, afr[i], false, bfr[j], (short)0, acc[i][j], false, false);
    }

    // ------------------------------ epilogue ------------------------------
    float delta = 0.f, zp = 0.f;
    if (EPI == EPI_FQ) { delta = *s_delta; zp = *s_zp; }

    #pragma unroll
    for (int i = 0; i < 2; ++i) {
        #pragma unroll
        for (int j = 0; j < 4; ++j) {
            int col = bn0 + wn * 64 + j * 16 + l16;
            int rowBase = bm0 + wm * 32 + i * 16 + hi * 8;
            #pragma unroll
            for (int r = 0; r < 8; ++r) {
                int row = rowBase + r;
                float v = acc[i][j][r] * alpha;
                if (EPI == EPI_F32) {
                    Cf[(size_t)row * ldc + col] = v;
                } else if (EPI == EPI_FQ) {
                    float q = rintf(v / delta) + zp;
                    q = fminf(fmaxf(q, 0.f), 255.f);
                    Ch[(size_t)row * ldc + col] = (_Float16)((q - zp) * delta);
                } else if (EPI == EPI_F16) {
                    Ch[(size_t)row * ldc + col] = (_Float16)v;
                } else if (EPI == EPI_BIAS_F16) {
                    Ch[(size_t)row * ldc + col] = (_Float16)(v + bias[col]);
                } else { // EPI_BIAS_RES_F32
                    Cf[(size_t)row * ldc + col] =
                        v + bias[col] + resid[(size_t)row * ldr + col];
                }
            }
        }
    }
}

// ------------------------------ P@V GEMM -----------------------------------
// C[M,64] (f16) = A[M,K] (f16, lda) @ B[K][N] (f16, ldb, K-major).
// Requires M%128==0, N==64, K%32==0. A via async DMA; B transposed in LDS.
__global__ __launch_bounds__(256) void k_gemm_pv(
    const _Float16* __restrict__ A, int lda,
    const _Float16* __restrict__ B, int ldb,
    _Float16* __restrict__ Ch, int ldc,
    int M, int K)
{
    __shared__ alignas(16) _Float16 As[2][BM][LDS_STRIDE];
    __shared__ alignas(16) _Float16 Bs[2][BNPV][LDS_STRIDE];

    const int tid  = threadIdx.x;
    const int lane = tid & 31;
    const int wid  = tid >> 5;
    const int l16  = lane & 15;
    const int hi   = (lane >> 4) & 1;
    const int wm   = wid & 3;            // 4 x 32 rows
    const int wn   = wid >> 2;           // 2 x 32 cols
    const int bm0  = blockIdx.x * BM;

    v8f acc[2][2] = {};

    const int ar  = tid >> 2;          // 0..63
    const int akc = (tid & 3) * 8;
    const int bkr = tid >> 3;          // 0..31 (k row)
    const int bnc = (tid & 7) * 8;     // n offset
    const _Float16* Arow0 = A + (size_t)(bm0 + ar) * lda + akc;
    const _Float16* Arow1 = A + (size_t)(bm0 + ar + 64) * lda + akc;
    const _Float16* Brow  = B + (size_t)bkr * ldb + bnc;
    const size_t bstep = (size_t)BK * ldb;

    unsigned lA0[2], lA1[2];
    #pragma unroll
    for (int s = 0; s < 2; ++s) {
        lA0[s] = lds_addr(&As[s][ar][akc]);
        lA1[s] = lds_addr(&As[s][ar + 64][akc]);
    }

    int4 pb;
    // prologue
    async_b128(lA0[0], Arow0);
    async_b128(lA1[0], Arow1);
    pb = *(const int4*)(Brow);
    {
        union { int4 q; _Float16 h[8]; } t; t.q = pb;
        #pragma unroll
        for (int j = 0; j < 8; ++j) Bs[0][bnc + j][bkr] = t.h[j];
    }
    wait_async0();
    __syncthreads();

    int buf = 0;
    for (int k0 = BK; k0 < K; k0 += BK) {
        int nb = buf ^ 1;
        async_b128(lA0[nb], Arow0 + k0);
        async_b128(lA1[nb], Arow1 + k0);
        pb = *(const int4*)(Brow + (size_t)(k0 / BK) * bstep);
        {
            v16h afr[2], bfr[2];
            #pragma unroll
            for (int i = 0; i < 2; ++i)
                afr[i] = load_frag(&As[buf][wm * 32 + i * 16 + l16][0], hi);
            #pragma unroll
            for (int j = 0; j < 2; ++j)
                bfr[j] = load_frag(&Bs[buf][wn * 32 + j * 16 + l16][0], hi);
            #pragma unroll
            for (int i = 0; i < 2; ++i)
                #pragma unroll
                for (int j = 0; j < 2; ++j)
                    acc[i][j] = __builtin_amdgcn_wmma_f32_16x16x32_f16(
                        false, afr[i], false, bfr[j], (short)0, acc[i][j], false, false);
        }
        {
            union { int4 q; _Float16 h[8]; } t; t.q = pb;
            #pragma unroll
            for (int j = 0; j < 8; ++j) Bs[nb][bnc + j][bkr] = t.h[j];
        }
        wait_async0();
        __syncthreads();
        buf = nb;
    }
    {
        v16h afr[2], bfr[2];
        #pragma unroll
        for (int i = 0; i < 2; ++i)
            afr[i] = load_frag(&As[buf][wm * 32 + i * 16 + l16][0], hi);
        #pragma unroll
        for (int j = 0; j < 2; ++j)
            bfr[j] = load_frag(&Bs[buf][wn * 32 + j * 16 + l16][0], hi);
        #pragma unroll
        for (int i = 0; i < 2; ++i)
            #pragma unroll
            for (int j = 0; j < 2; ++j)
                acc[i][j] = __builtin_amdgcn_wmma_f32_16x16x32_f16(
                    false, afr[i], false, bfr[j], (short)0, acc[i][j], false, false);
    }

    #pragma unroll
    for (int i = 0; i < 2; ++i) {
        #pragma unroll
        for (int j = 0; j < 2; ++j) {
            int col = wn * 32 + j * 16 + l16;
            int rowBase = bm0 + wm * 32 + i * 16 + hi * 8;
            #pragma unroll
            for (int r = 0; r < 8; ++r)
                Ch[(size_t)(rowBase + r) * ldc + col] = (_Float16)acc[i][j][r];
        }
    }
}

// --------------------------- LayerNorm (row=1024) --------------------------
__global__ __launch_bounds__(256) void k_layernorm(
    const float* __restrict__ X, const float* __restrict__ w,
    const float* __restrict__ b, _Float16* __restrict__ Y, int D)
{
    __shared__ float red[16];
    const int row = blockIdx.x;
    const float* x = X + (size_t)row * D;
    float s = 0.f, ss = 0.f;
    for (int i = threadIdx.x; i < D; i += 256) { float v = x[i]; s += v; ss += v * v; }
    for (int o = 16; o > 0; o >>= 1) { s += __shfl_down(s, o); ss += __shfl_down(ss, o); }
    int lane = threadIdx.x & 31, wid = threadIdx.x >> 5;
    if (lane == 0) { red[wid] = s; red[8 + wid] = ss; }
    __syncthreads();
    if (wid == 0) {
        s  = (lane < 8) ? red[lane] : 0.f;
        ss = (lane < 8) ? red[8 + lane] : 0.f;
        for (int o = 4; o > 0; o >>= 1) { s += __shfl_down(s, o); ss += __shfl_down(ss, o); }
        if (lane == 0) { red[0] = s; red[1] = ss; }
    }
    __syncthreads();
    float mu  = red[0] / D;
    float var = red[1] / D - mu * mu;
    float rs  = rsqrtf(var + 1e-5f);
    _Float16* y = Y + (size_t)row * D;
    for (int i = threadIdx.x; i < D; i += 256)
        y[i] = (_Float16)((x[i] - mu) * rs * w[i] + b[i]);
}

// ---------------- softmax over row + always_zero fake-quant ----------------
__global__ __launch_bounds__(256) void k_softmax_fq(
    const float* __restrict__ S, int lds_, _Float16* __restrict__ P, int ldp,
    int mlen, const float* __restrict__ s_dw)
{
    __shared__ float red[8];
    const int row = blockIdx.x;
    const float* s = S + (size_t)row * lds_;
    int lane = threadIdx.x & 31, wid = threadIdx.x >> 5;

    float mx = -3.0e38f;
    for (int i = threadIdx.x; i < mlen; i += 256) mx = fmaxf(mx, s[i]);
    for (int o = 16; o > 0; o >>= 1) mx = fmaxf(mx, __shfl_down(mx, o));
    if (lane == 0) red[wid] = mx;
    __syncthreads();
    if (wid == 0) {
        mx = (lane < 8) ? red[lane] : -3.0e38f;
        for (int o = 4; o > 0; o >>= 1) mx = fmaxf(mx, __shfl_down(mx, o));
        if (lane == 0) red[0] = mx;
    }
    __syncthreads();
    mx = red[0];
    __syncthreads();

    float sum = 0.f;
    for (int i = threadIdx.x; i < mlen; i += 256) sum += expf(s[i] - mx);
    for (int o = 16; o > 0; o >>= 1) sum += __shfl_down(sum, o);
    if (lane == 0) red[wid] = sum;
    __syncthreads();
    if (wid == 0) {
        sum = (lane < 8) ? red[lane] : 0.f;
        for (int o = 4; o > 0; o >>= 1) sum += __shfl_down(sum, o);
        if (lane == 0) red[0] = sum;
    }
    __syncthreads();
    float inv = 1.f / red[0];
    float dw  = *s_dw;

    _Float16* p = P + (size_t)row * ldp;
    for (int i = threadIdx.x; i < ldp; i += 256) {
        if (i < mlen) {
            float pv = expf(s[i] - mx) * inv;
            float q  = fminf(fmaxf(rintf(pv / dw), 0.f), 255.f);
            p[i] = (_Float16)(q * dw);
        } else {
            p[i] = (_Float16)0.f;
        }
    }
}

// ---------------- transpose + f32->f16 convert (W[K][N] -> Wt[N][K]) -------
__global__ __launch_bounds__(256) void k_transpose_cvt(
    const float* __restrict__ W, _Float16* __restrict__ Wt, int K, int N)
{
    __shared__ float t[32][33];
    int bx = blockIdx.x * 32;  // over N
    int by = blockIdx.y * 32;  // over K
    int tx = threadIdx.x & 31, ty = threadIdx.x >> 5;  // 8 rows/pass
    #pragma unroll
    for (int p = 0; p < 4; ++p) {
        int k = by + ty + p * 8, n = bx + tx;
        t[ty + p * 8][tx] = (k < K && n < N) ? W[(size_t)k * N + n] : 0.f;
    }
    __syncthreads();
    #pragma unroll
    for (int p = 0; p < 4; ++p) {
        int n = bx + ty + p * 8, k = by + tx;
        if (n < N && k < K) Wt[(size_t)n * K + k] = (_Float16)t[tx][ty + p * 8];
    }
}

// ---------- context f32 -> f16 with per-batch row padding (77 -> 128) ------
__global__ __launch_bounds__(256) void k_cvt_pad(const float* __restrict__ X,
                                                 _Float16* __restrict__ Y,
                                                 int padRows, int realRows, int C,
                                                 int total)
{
    int i = blockIdx.x * 256 + threadIdx.x;
    if (i >= total) return;
    int r  = i / C;
    int c  = i - r * C;
    int b  = r / padRows;
    int rr = r - b * padRows;
    Y[i] = (rr < realRows) ? (_Float16)X[((size_t)b * realRows + rr) * C + c]
                           : (_Float16)0.f;
}

// ----------------------------- GEGLU (tanh gelu) ---------------------------
__global__ __launch_bounds__(256) void k_geglu(const _Float16* __restrict__ Pj,
                                               _Float16* __restrict__ O, int rows)
{
    size_t i = (size_t)blockIdx.x * 256 + threadIdx.x;
    size_t total = (size_t)rows * 4096;
    if (i >= total) return;
    size_t r = i >> 12, c = i & 4095;
    float a = (float)Pj[r * 8192 + c];
    float g = (float)Pj[r * 8192 + 4096 + c];
    float t = tanhf(0.7978845608028654f * (g + 0.044715f * g * g * g));
    O[i] = (_Float16)(a * 0.5f * g * (1.f + t));
}

// ------------------------------ host dispatch ------------------------------
static inline void launch_gemm(hipStream_t st, int epi,
    const _Float16* A, int lda, const _Float16* B, int ldb,
    float* Cf, _Float16* Ch, int ldc, int M, int N, int K,
    const float* bias, const float* resid, int ldr,
    const float* sd, const float* sz, float alpha)
{
    dim3 g(M / BM, N / BN), b(256);
    switch (epi) {
    case EPI_F32:          k_gemm<EPI_F32><<<g, b, 0, st>>>(A, lda, B, ldb, Cf, Ch, ldc, M, N, K, bias, resid, ldr, sd, sz, alpha); break;
    case EPI_FQ:           k_gemm<EPI_FQ><<<g, b, 0, st>>>(A, lda, B, ldb, Cf, Ch, ldc, M, N, K, bias, resid, ldr, sd, sz, alpha); break;
    case EPI_F16:          k_gemm<EPI_F16><<<g, b, 0, st>>>(A, lda, B, ldb, Cf, Ch, ldc, M, N, K, bias, resid, ldr, sd, sz, alpha); break;
    case EPI_BIAS_F16:     k_gemm<EPI_BIAS_F16><<<g, b, 0, st>>>(A, lda, B, ldb, Cf, Ch, ldc, M, N, K, bias, resid, ldr, sd, sz, alpha); break;
    default:               k_gemm<EPI_BIAS_RES_F32><<<g, b, 0, st>>>(A, lda, B, ldb, Cf, Ch, ldc, M, N, K, bias, resid, ldr, sd, sz, alpha); break;
    }
}

extern "C" void kernel_launch(void* const* d_in, const int* in_sizes, int n_in,
                              void* d_out, int out_size, void* d_ws, size_t ws_size,
                              hipStream_t stream)
{
    // problem dims
    const int Bb = 2, Nn = 2048, Dd = 1024, Hh = 16, DH = 64, Mm = 77, CD = 768, FF = 4096;
    const int MP = 128;              // padded context rows per batch
    const int MN = Bb * Nn;          // 4096 token rows
    const float alpha = 0.125f;      // DH^-0.5

    // ---- inputs (setup_inputs order) ----
    const float* x     = (const float*)d_in[0];
    const float* ctx   = (const float*)d_in[1];
    const float* n1w   = (const float*)d_in[2];
    const float* n1b   = (const float*)d_in[3];
    const float* n2w   = (const float*)d_in[4];
    const float* n2b   = (const float*)d_in[5];
    const float* n3w   = (const float*)d_in[6];
    const float* n3b   = (const float*)d_in[7];
    const float* a1_wq = (const float*)d_in[8];
    const float* a1_wk = (const float*)d_in[9];
    const float* a1_wv = (const float*)d_in[10];
    const float* a1_wo = (const float*)d_in[11];
    const float* a1_bo = (const float*)d_in[12];
    const float* a1_dq = (const float*)d_in[13];
    const float* a1_zq = (const float*)d_in[14];
    const float* a1_dk = (const float*)d_in[15];
    const float* a1_zk = (const float*)d_in[16];
    const float* a1_dv = (const float*)d_in[17];
    const float* a1_zv = (const float*)d_in[18];
    const float* a1_dw = (const float*)d_in[19];
    const float* a2_wq = (const float*)d_in[20];
    const float* a2_wk = (const float*)d_in[21];
    const float* a2_wv = (const float*)d_in[22];
    const float* a2_wo = (const float*)d_in[23];
    const float* a2_bo = (const float*)d_in[24];
    const float* a2_dq = (const float*)d_in[25];
    const float* a2_zq = (const float*)d_in[26];
    const float* a2_dk = (const float*)d_in[27];
    const float* a2_zk = (const float*)d_in[28];
    const float* a2_dv = (const float*)d_in[29];
    const float* a2_zv = (const float*)d_in[30];
    const float* a2_dw = (const float*)d_in[31];
    const float* ff_w1 = (const float*)d_in[32];
    const float* ff_b1 = (const float*)d_in[33];
    const float* ff_w2 = (const float*)d_in[34];
    const float* ff_b2 = (const float*)d_in[35];
    float* out = (float*)d_out;
    (void)in_sizes; (void)n_in; (void)out_size; (void)ws_size;

    // ---- workspace carve ----
    char* wsp = (char*)d_ws;
    size_t off = 0;
    auto alloc = [&](size_t bytes) -> void* {
        void* p = wsp + off;
        off += (bytes + 255) & ~(size_t)255;
        return p;
    };
    _Float16* wq1t  = (_Float16*)alloc((size_t)Dd * Dd * 2);
    _Float16* wk1t  = (_Float16*)alloc((size_t)Dd * Dd * 2);
    _Float16* wv1t  = (_Float16*)alloc((size_t)Dd * Dd * 2);
    _Float16* wo1t  = (_Float16*)alloc((size_t)Dd * Dd * 2);
    _Float16* wq2t  = (_Float16*)alloc((size_t)Dd * Dd * 2);
    _Float16* wk2t  = (_Float16*)alloc((size_t)Dd * CD * 2);
    _Float16* wv2t  = (_Float16*)alloc((size_t)Dd * CD * 2);
    _Float16* wo2t  = (_Float16*)alloc((size_t)Dd * Dd * 2);
    _Float16* w1t   = (_Float16*)alloc((size_t)(2 * FF) * Dd * 2);
    _Float16* w2t   = (_Float16*)alloc((size_t)Dd * FF * 2);
    _Float16* ctx16p= (_Float16*)alloc((size_t)Bb * MP * CD * 2);   // padded ctx
    _Float16* hln   = (_Float16*)alloc((size_t)MN * Dd * 2);
    _Float16* qf    = (_Float16*)alloc((size_t)MN * Dd * 2);
    _Float16* kf    = (_Float16*)alloc((size_t)MN * Dd * 2);
    _Float16* vf    = (_Float16*)alloc((size_t)MN * Dd * 2);
    _Float16* kf2   = (_Float16*)alloc((size_t)Bb * MP * Dd * 2);   // padded K2
    _Float16* vf2   = (_Float16*)alloc((size_t)Bb * MP * Dd * 2);   // padded V2
    float*    sim   = (float*)alloc((size_t)Nn * Nn * 4);
    _Float16* probs = (_Float16*)alloc((size_t)Nn * Nn * 2);
    _Float16* attno = (_Float16*)alloc((size_t)MN * Dd * 2);
    float*    x1    = (float*)alloc((size_t)MN * Dd * 4);
    float*    x2    = (float*)alloc((size_t)MN * Dd * 4);
    _Float16* proj  = (_Float16*)alloc((size_t)MN * 2 * FF * 2);
    _Float16* gg    = (_Float16*)alloc((size_t)MN * FF * 2);

    dim3 tb(256);
    auto tgrid = [](int K, int N) { return dim3((N + 31) / 32, (K + 31) / 32); };

    // ---- weight prep: W[K][N] f32 -> Wt[N][K] f16 ----
    k_transpose_cvt<<<tgrid(Dd, Dd), tb, 0, stream>>>(a1_wq, wq1t, Dd, Dd);
    k_transpose_cvt<<<tgrid(Dd, Dd), tb, 0, stream>>>(a1_wk, wk1t, Dd, Dd);
    k_transpose_cvt<<<tgrid(Dd, Dd), tb, 0, stream>>>(a1_wv, wv1t, Dd, Dd);
    k_transpose_cvt<<<tgrid(Dd, Dd), tb, 0, stream>>>(a1_wo, wo1t, Dd, Dd);
    k_transpose_cvt<<<tgrid(Dd, Dd), tb, 0, stream>>>(a2_wq, wq2t, Dd, Dd);
    k_transpose_cvt<<<tgrid(CD, Dd), tb, 0, stream>>>(a2_wk, wk2t, CD, Dd);
    k_transpose_cvt<<<tgrid(CD, Dd), tb, 0, stream>>>(a2_wv, wv2t, CD, Dd);
    k_transpose_cvt<<<tgrid(Dd, Dd), tb, 0, stream>>>(a2_wo, wo2t, Dd, Dd);
    k_transpose_cvt<<<tgrid(Dd, 2 * FF), tb, 0, stream>>>(ff_w1, w1t, Dd, 2 * FF);
    k_transpose_cvt<<<tgrid(FF, Dd), tb, 0, stream>>>(ff_w2, w2t, FF, Dd);
    {
        int total = Bb * MP * CD;
        k_cvt_pad<<<(total + 255) / 256, tb, 0, stream>>>(ctx, ctx16p, MP, Mm, CD, total);
    }

    // ==================== self-attention ====================
    k_layernorm<<<MN, tb, 0, stream>>>(x, n1w, n1b, hln, Dd);

    launch_gemm(stream, EPI_FQ, hln, Dd, wq1t, Dd, nullptr, qf, Dd, MN, Dd, Dd, nullptr, nullptr, 0, a1_dq, a1_zq, 1.f);
    launch_gemm(stream, EPI_FQ, hln, Dd, wk1t, Dd, nullptr, kf, Dd, MN, Dd, Dd, nullptr, nullptr, 0, a1_dk, a1_zk, 1.f);
    launch_gemm(stream, EPI_FQ, hln, Dd, wv1t, Dd, nullptr, vf, Dd, MN, Dd, Dd, nullptr, nullptr, 0, a1_dv, a1_zv, 1.f);

    for (int bh = 0; bh < Bb * Hh; ++bh) {
        int b = bh >> 4, h = bh & 15;
        const _Float16* Qp = qf + (size_t)b * Nn * Dd + h * DH;
        const _Float16* Kp = kf + (size_t)b * Nn * Dd + h * DH;
        const _Float16* Vp = vf + (size_t)b * Nn * Dd + h * DH;
        _Float16*       Op = attno + (size_t)b * Nn * Dd + h * DH;
        // sim = (Q @ K^T) * alpha   (K rows are already the Bt form)
        launch_gemm(stream, EPI_F32, Qp, Dd, Kp, Dd, sim, nullptr, Nn, Nn, Nn, DH, nullptr, nullptr, 0, nullptr, nullptr, alpha);
        k_softmax_fq<<<Nn, tb, 0, stream>>>(sim, Nn, probs, Nn, Nn, a1_dw);
        // out = probs @ V   (V is [K=m][N=d] -> K-major staging)
        k_gemm_pv<<<dim3(Nn / BM, 1), tb, 0, stream>>>(probs, Nn, Vp, Dd, Op, Dd, Nn, Nn);
    }
    // x1 = attno @ wo1 + bo + x
    launch_gemm(stream, EPI_BIAS_RES_F32, attno, Dd, wo1t, Dd, x1, nullptr, Dd, MN, Dd, Dd, a1_bo, x, Dd, nullptr, nullptr, 1.f);

    // ==================== cross-attention ====================
    k_layernorm<<<MN, tb, 0, stream>>>(x1, n2w, n2b, hln, Dd);

    launch_gemm(stream, EPI_FQ, hln, Dd, wq2t, Dd, nullptr, qf, Dd, MN, Dd, Dd, nullptr, nullptr, 0, a2_dq, a2_zq, 1.f);
    // padded K2/V2 projections: M = Bb*128 (zero rows stay zero through fq)
    launch_gemm(stream, EPI_FQ, ctx16p, CD, wk2t, CD, nullptr, kf2, Dd, Bb * MP, Dd, CD, nullptr, nullptr, 0, a2_dk, a2_zk, 1.f);
    launch_gemm(stream, EPI_FQ, ctx16p, CD, wv2t, CD, nullptr, vf2, Dd, Bb * MP, Dd, CD, nullptr, nullptr, 0, a2_dv, a2_zv, 1.f);

    for (int bh = 0; bh < Bb * Hh; ++bh) {
        int b = bh >> 4, h = bh & 15;
        const _Float16* Qp = qf + (size_t)b * Nn * Dd + h * DH;
        const _Float16* Kp = kf2 + (size_t)b * MP * Dd + h * DH;
        const _Float16* Vp = vf2 + (size_t)b * MP * Dd + h * DH;
        _Float16*       Op = attno + (size_t)b * Nn * Dd + h * DH;
        // sim over padded 128 keys; softmax masks to the real 77
        launch_gemm(stream, EPI_F32, Qp, Dd, Kp, Dd, sim, nullptr, MP, Nn, MP, DH, nullptr, nullptr, 0, nullptr, nullptr, alpha);
        k_softmax_fq<<<Nn, tb, 0, stream>>>(sim, MP, probs, MP, Mm, a2_dw);
        // P@V with K=128; pad rows of V are zero and pad probs are zero
        k_gemm_pv<<<dim3(Nn / BM, 1), tb, 0, stream>>>(probs, MP, Vp, Dd, Op, Dd, Nn, MP);
    }
    // x2 = attno @ wo2 + bo + x1
    launch_gemm(stream, EPI_BIAS_RES_F32, attno, Dd, wo2t, Dd, x2, nullptr, Dd, MN, Dd, Dd, a2_bo, x1, Dd, nullptr, nullptr, 1.f);

    // ==================== GEGLU feed-forward ====================
    k_layernorm<<<MN, tb, 0, stream>>>(x2, n3w, n3b, hln, Dd);

    // proj = hln @ ff_w1 + b1  (f16, [MN][8192])
    launch_gemm(stream, EPI_BIAS_F16, hln, Dd, w1t, Dd, nullptr, proj, 2 * FF, MN, 2 * FF, Dd, ff_b1, nullptr, 0, nullptr, nullptr, 1.f);
    {
        size_t total = (size_t)MN * FF;
        k_geglu<<<(unsigned)((total + 255) / 256), tb, 0, stream>>>(proj, gg, MN);
    }
    // out = gg @ ff_w2 + b2 + x2
    launch_gemm(stream, EPI_BIAS_RES_F32, gg, FF, w2t, FF, out, nullptr, Dd, MN, Dd, FF, ff_b2, x2, Dd, nullptr, nullptr, 1.f);
}
